// Attention_34737695490195
// MI455X (gfx1250) — matrix operand order repaired and checked by reading, not compiled, not run on hardware
//
#include <hip/hip_runtime.h>

// Problem dims (match reference)
constexpr int Bb = 4;
constexpr int S  = 4096;
constexpr int D  = 512;
constexpr int Hh = 64;
constexpr float QK_SCALE = 0.125f; // 1/sqrt(64)

typedef __attribute__((ext_vector_type(16))) _Float16 v16h;
typedef __attribute__((ext_vector_type(8)))  float    v8f;
typedef __attribute__((ext_vector_type(4)))  int      v4i;

union AFrag { v16h v; _Float16 h[16]; };
union CFrag { v8f  v; float    f[8];  };

// CDNA5 async memory->LDS path (builtin verified present on this toolchain;
// parameter types are v4i32 pointers in addrspace(1)/addrspace(3)).
#if defined(__gfx1250__) && __has_builtin(__builtin_amdgcn_global_load_async_to_lds_b128)
#define HAVE_ASYNC_LDS 1
typedef __attribute__((address_space(1))) v4i as1_v4i;
typedef __attribute__((address_space(3))) v4i as3_v4i;
#endif

__device__ inline void wait_asynccnt0() {
#if __has_builtin(__builtin_amdgcn_s_wait_asynccnt)
    __builtin_amdgcn_s_wait_asynccnt(0);
#elif defined(__gfx1250__)
    asm volatile("s_wait_asynccnt 0" ::: "memory");
#endif
}

// 16B memory->LDS copy: async DMA on CDNA5 (tracked by ASYNCcnt, no VGPR
// round trip), plain vector copy otherwise.
__device__ inline void cp16_to_lds(_Float16* dst, const _Float16* src) {
#if defined(HAVE_ASYNC_LDS)
    __builtin_amdgcn_global_load_async_to_lds_b128(
        (as1_v4i*)src, (as3_v4i*)dst, 0, 0);
#else
    *(uint4*)dst = *(const uint4*)src;
#endif
}

// ---- fragment load helpers -------------------------------------------------
// 16-bit A layout (16x32, ISA 7.12.2): lane&15 = M; elements 0..7 hold
// K = 8*(lane>>4)+e, elements 8..15 hold K = 16 + 8*(lane>>4) + (e-8).
__device__ inline void ld8_f32_as_h(AFrag& f, int idx, const float* p) {
    const float4 u0 = *(const float4*)(p + 0);
    const float4 u1 = *(const float4*)(p + 4);
    f.h[idx+0] = (_Float16)u0.x; f.h[idx+1] = (_Float16)u0.y;
    f.h[idx+2] = (_Float16)u0.z; f.h[idx+3] = (_Float16)u0.w;
    f.h[idx+4] = (_Float16)u1.x; f.h[idx+5] = (_Float16)u1.y;
    f.h[idx+6] = (_Float16)u1.z; f.h[idx+7] = (_Float16)u1.w;
}
__device__ inline void ld8_h(AFrag& f, int idx, const _Float16* p) {
    uint4 u = *(const uint4*)p;
    const _Float16* s = (const _Float16*)&u;
    #pragma unroll
    for (int j = 0; j < 8; ++j) f.h[idx+j] = s[j];
}

// ============================================================================
// Kernel 0: one-shot f32 -> f16 conversion of Wq|Wk|Wv (3*D*H elements).
// ============================================================================
__global__ __launch_bounds__(256)
void wcvt_kernel(const float* __restrict__ Wq, const float* __restrict__ Wk,
                 const float* __restrict__ Wv, _Float16* __restrict__ w16)
{
    const int f = (blockIdx.x * 256 + threadIdx.x) * 4;   // 4 elems / thread
    const int m   = f / (D * Hh);
    const int off = f % (D * Hh);
    const float* srcs[3] = { Wq, Wk, Wv };
    const float4 u = *(const float4*)(srcs[m] + off);
    union { uint2 u2; _Float16 h[4]; } r;
    r.h[0] = (_Float16)u.x; r.h[1] = (_Float16)u.y;
    r.h[2] = (_Float16)u.z; r.h[3] = (_Float16)u.w;
    *(uint2*)(w16 + f) = r.u2;
}

// ============================================================================
// Kernel 1: fused QKV projection.  x:[B*S, D] f32 * W16:[D, H] f16 (+bias) ->
// q,k,v stored f16 (q pre-scaled by 1/sqrt(H)).  One wave owns a 16-row tile.
// ============================================================================
__global__ __launch_bounds__(256)
void qkv_kernel(const float* __restrict__ x, const _Float16* __restrict__ w16,
                const float* __restrict__ bq, const float* __restrict__ bk,
                const float* __restrict__ bv,
                _Float16* __restrict__ qo, _Float16* __restrict__ ko,
                _Float16* __restrict__ vo)
{
    const int lane = threadIdx.x & 31;
    const int wave = threadIdx.x >> 5;
    const int row0 = (blockIdx.x * 8 + wave) * 16;   // tile over flattened B*S
    const int mrow = lane & 15;
    const int hi   = lane >> 4;

    CFrag acc[3][4];
    #pragma unroll
    for (int m = 0; m < 3; ++m)
        #pragma unroll
        for (int c = 0; c < 4; ++c)
            #pragma unroll
            for (int r = 0; r < 8; ++r) acc[m][c].f[r] = 0.0f;

    for (int kc = 0; kc < D / 32; ++kc) {
        // A fragment of x (16x32), f32 -> f16 (x is used once: convert inline)
        AFrag a;
        const float* xb = x + (size_t)(row0 + mrow) * D + kc * 32 + hi * 8;
        ld8_f32_as_h(a, 0, xb);
        ld8_f32_as_h(a, 8, xb + 16);

        #pragma unroll
        for (int m = 0; m < 3; ++m) {
            // B layout (32x16): lane&15 = N column, elements hold K in A's table order.
            const _Float16* Wb = w16 + (size_t)m * D * Hh
                                     + (size_t)(kc * 32) * Hh;
            #pragma unroll
            for (int c = 0; c < 4; ++c) {
                AFrag b;
                #pragma unroll
                for (int e = 0; e < 16; ++e)
                    b.h[e] = Wb[(size_t)(8 * hi + e + (e & 8)) * Hh + c * 16 + mrow];
                acc[m][c].v = __builtin_amdgcn_wmma_f32_16x16x32_f16(
                    false, a.v, false, b.v, (short)0, acc[m][c].v, false, false);
            }
        }
    }

    _Float16* outs[3]     = { qo, ko, vo };
    const float* bias3[3] = { bq, bk, bv };
    #pragma unroll
    for (int m = 0; m < 3; ++m) {
        #pragma unroll
        for (int c = 0; c < 4; ++c) {
            const int n = c * 16 + (lane & 15);   // C layout: lane&15 = column
            const float bias = bias3[m][n];
            #pragma unroll
            for (int r = 0; r < 8; ++r) {
                float val = acc[m][c].f[r] + bias;
                if (m == 0) val *= QK_SCALE;      // fold 1/sqrt(H) into q
                outs[m][(size_t)(row0 + r + hi * 8) * Hh + n] = (_Float16)val;
            }
        }
    }
}

// ============================================================================
// Kernel 2: flash attention, transposed-score formulation.
//   S^T = K_tile (A) x Q^T (B)  =>  C-layout of S^T is element-identical to
//   the A-layout needed for P in P x V: the softmax transpose is free in
//   registers (no LDS round trip).  K/V tiles double-buffered via
//   global_load_async_to_lds_b128; one barrier + one s_wait_asynccnt per
//   32-key block.  Block = 4 waves = 64 queries of one batch.
// ============================================================================
__global__ __launch_bounds__(128)
void attn_kernel(const _Float16* __restrict__ q,
                 const _Float16* __restrict__ k,
                 const _Float16* __restrict__ v,
                 float* __restrict__ out)
{
    __shared__ alignas(16) _Float16 kS[2][32 * 64];   // K block [key][h], x2 buf
    __shared__ alignas(16) _Float16 vS[2][32 * 64];   // V block [key][h], x2 buf

    const int t     = threadIdx.x;          // 0..127
    const int lane  = t & 31;
    const int wave  = t >> 5;               // 0..3
    const int batch = blockIdx.x / (S / 64);
    const int q0    = (blockIdx.x % (S / 64)) * 64 + wave * 16;
    const int nq    = lane & 15;            // this lane's query (columns of S^T C)
    const int hi    = lane >> 4;

    // Q^T B fragments (loop-invariant): B is 32(h) x 16(query); lane&15 = query column,
    // 16 h values per lane (strided gather, done once per wave).
    AFrag bQ[2];
    #pragma unroll
    for (int hk = 0; hk < 2; ++hk)
        #pragma unroll
        for (int e = 0; e < 16; ++e)
            bQ[hk].h[e] = q[(size_t)(batch * S + q0 + nq) * Hh + hk * 32 + 8 * hi + e + (e & 8)];

    CFrag o[4];
    #pragma unroll
    for (int c = 0; c < 4; ++c)
        #pragma unroll
        for (int r = 0; r < 8; ++r) o[c].f[r] = 0.0f;
    float m_n = -1e30f, l_n = 0.0f;         // per-lane stats for query nq

    const _Float16* kbase = k + (size_t)batch * S * Hh;
    const _Float16* vbase = v + (size_t)batch * S * Hh;

    // Stage one 32x64 f16 tile (4 KB): 128 threads x 2 x 16B each.
    auto issue_tile = [&](int kb, int buf) {
        const size_t g = (size_t)kb * 32 * Hh;
        cp16_to_lds(kS[buf] + t * 8,        kbase + g + t * 8);
        cp16_to_lds(kS[buf] + 1024 + t * 8, kbase + g + 1024 + t * 8);
        cp16_to_lds(vS[buf] + t * 8,        vbase + g + t * 8);
        cp16_to_lds(vS[buf] + 1024 + t * 8, vbase + g + 1024 + t * 8);
    };

    issue_tile(0, 0);
    for (int kb = 0; kb < S / 32; ++kb) {
        wait_asynccnt0();        // this wave's portion of tile kb landed
        __syncthreads();         // all waves' portions landed; prev reads done
        if (kb + 1 < S / 32) issue_tile(kb + 1, (kb + 1) & 1);

        const _Float16* kB = kS[kb & 1];
        const _Float16* vB = vS[kb & 1];

        // ---- S^T (2 sub-tiles of 16 keys): A = K rows (contiguous in LDS)
        CFrag s0, s1;
        #pragma unroll
        for (int r = 0; r < 8; ++r) { s0.f[r] = 0.0f; s1.f[r] = 0.0f; }
        #pragma unroll
        for (int hk = 0; hk < 2; ++hk) {
            AFrag ak0, ak1;
            const _Float16* kp0 = kB + nq * 64 + hk * 32 + hi * 8;
            const _Float16* kp1 = kp0 + 16 * 64;
            ld8_h(ak0, 0, kp0); ld8_h(ak0, 8, kp0 + 16);
            ld8_h(ak1, 0, kp1); ld8_h(ak1, 8, kp1 + 16);
            s0.v = __builtin_amdgcn_wmma_f32_16x16x32_f16(
                false, ak0.v, false, bQ[hk].v, (short)0, s0.v, false, false);
            s1.v = __builtin_amdgcn_wmma_f32_16x16x32_f16(
                false, ak1.v, false, bQ[hk].v, (short)0, s1.v, false, false);
        }

        // ---- online softmax: lane owns query nq; its 16 elements cover keys
        // {8*hi+0..7, 16+8*hi+0..7}; lane^16 holds the complementary 16.
        float rm = fmaxf(s0.f[0], s1.f[0]);
        #pragma unroll
        for (int r = 1; r < 8; ++r) rm = fmaxf(rm, fmaxf(s0.f[r], s1.f[r]));
        rm = fmaxf(rm, __shfl_xor(rm, 16, 32));
        const float mn   = fmaxf(m_n, rm);
        const float corr = __expf(m_n - mn);
        float rs = 0.0f;
        #pragma unroll
        for (int r = 0; r < 8; ++r) {
            s0.f[r] = __expf(s0.f[r] - mn);
            s1.f[r] = __expf(s1.f[r] - mn);
            rs += s0.f[r] + s1.f[r];
        }
        rs += __shfl_xor(rs, 16, 32);
        l_n = l_n * corr + rs;
        m_n = mn;

        // ---- free transpose: S^T C-layout == P A-layout, element for element
        AFrag pa;
        #pragma unroll
        for (int e = 0; e < 8; ++e) {
            pa.h[e]     = (_Float16)s0.f[e];
            pa.h[8 + e] = (_Float16)s1.f[e];
        }

        // ---- rescale O rows (query r+8*hi; corr lives in lane (r+8*hi))
        #pragma unroll
        for (int r = 0; r < 8; ++r) {
            const float cr = __shfl(corr, r + hi * 8, 32);
            #pragma unroll
            for (int c = 0; c < 4; ++c) o[c].f[r] *= cr;
        }

        // ---- O += P (16q x 32k) * V_blk (32k x 64h)
        #pragma unroll
        for (int c = 0; c < 4; ++c) {
            AFrag bv_;                       // B layout: lane&15 = h column, elements = keys (table order)
            #pragma unroll
            for (int e = 0; e < 16; ++e)
                bv_.h[e] = vB[(8 * hi + e + (e & 8)) * 64 + c * 16 + nq];
            o[c].v = __builtin_amdgcn_wmma_f32_16x16x32_f16(
                false, pa.v, false, bv_.v, (short)0, o[c].v, false, false);
        }
    }

    // ---- epilogue: out = O / l, fp32.  l for query r+8*hi lives in that lane.
    float rinv[8];
    #pragma unroll
    for (int r = 0; r < 8; ++r) rinv[r] = 1.0f / __shfl(l_n, r + hi * 8, 32);
    float* ob = out + (size_t)(batch * S + q0) * Hh;
    #pragma unroll
    for (int c = 0; c < 4; ++c) {
        const int n = c * 16 + (lane & 15);   // O C-layout: lane&15 = h column
        #pragma unroll
        for (int r = 0; r < 8; ++r)
            ob[(size_t)(r + hi * 8) * Hh + n] = o[c].f[r] * rinv[r];
    }
}

// ============================================================================
extern "C" void kernel_launch(void* const* d_in, const int* in_sizes, int n_in,
                              void* d_out, int out_size, void* d_ws, size_t ws_size,
                              hipStream_t stream)
{
    const float* x  = (const float*)d_in[0];
    const float* Wq = (const float*)d_in[1];
    const float* bq = (const float*)d_in[2];
    const float* Wk = (const float*)d_in[3];
    const float* bk = (const float*)d_in[4];
    const float* Wv = (const float*)d_in[5];
    const float* bv = (const float*)d_in[6];

    // Workspace: q,k,v f16 [B*S, H] (2 MB apiece) + W f16 [3, D*H] (192 KB)
    _Float16* qws = (_Float16*)d_ws;
    _Float16* kws = qws + (size_t)Bb * S * Hh;
    _Float16* vws = kws + (size_t)Bb * S * Hh;
    _Float16* w16 = vws + (size_t)Bb * S * Hh;

    // Kernel 0: convert weights once (3*D*H / (256*4) = 96 blocks)
    wcvt_kernel<<<dim3(3 * D * Hh / (256 * 4)), dim3(256), 0, stream>>>(
        Wq, Wk, Wv, w16);

    // Kernel 1: 1024 wave-tiles of 16 rows, 8 waves / 256-thread block
    qkv_kernel<<<dim3((Bb * S / 16) / 8), dim3(256), 0, stream>>>(
        x, w16, bq, bk, bv, qws, kws, vws);

    // Kernel 2: 64 queries per block (4 waves x 16), 256 blocks
    attn_kernel<<<dim3(Bb * S / 64), dim3(128), 0, stream>>>(
        qws, kws, vws, (float*)d_out);
}
